// DeepRNN_49366354100497
// MI455X (gfx1250) — matrix operand order repaired and checked by reading, not compiled, not run on hardware
//
#include <hip/hip_runtime.h>
#include <stdint.h>

typedef unsigned short u16;

#define BB   64
#define TT   512
#define INN  512
#define HH   1024
#define OUTN 4096
#define BT   (BB*TT)
#define G4H  (4*HH)

// LDS tile geometry (u16 units): rows padded 32 -> 40 to break bank conflicts
#define ROWW   40                 // u16 per padded row
#define ROWB   80                 // bytes per padded row

typedef __attribute__((ext_vector_type(16))) __bf16 v16bf;
typedef __attribute__((ext_vector_type(8)))  float  v8f;

union FragBF { uint4 q[2]; v16bf v; };
union FragC  { float f[8]; v8f  v; };

__device__ __forceinline__ u16 f2bf(float f) {
  uint32_t u = __float_as_uint(f);
  u += 0x7FFFu + ((u >> 16) & 1u);   // round-to-nearest-even
  return (u16)(u >> 16);
}

__device__ __forceinline__ float sigm(float x) {
  return 1.0f / (1.0f + __expf(-x));
}

// CDNA5 async copy: global -> LDS, no VGPR round trip, tracked by ASYNCcnt.
__device__ __forceinline__ void async_b128(unsigned lds_off, const void* gaddr) {
  asm volatile("global_load_async_to_lds_b128 %0, %1, off"
               :: "v"(lds_off), "v"(gaddr) : "memory");
}
__device__ __forceinline__ void wait_async0() {
  asm volatile("s_wait_asynccnt 0" ::: "memory");
}

// ---------------------------------------------------------------------------
// bf16 WMMA GEMM:  C[M,N](f32) = A[M,K](bf16,rm) * B[N,K](bf16,rm)^T
//                                (+ bias1[n]) (+ bias2[n])
//                                (+ addC[m*addld + addoff + n])
// Block: 256 threads (8 waves), (64*MPW)x128 output tile, K stepped by 32,
// LDS double-buffered via global_load_async_to_lds_b128.
//   MPW = m-tiles per wave.  MPW=1 -> 64x128 tile (recurrent step, M=64)
//                            MPW=2 -> 128x128 tile (big GEMMs)
// grid = (N/128, M/(64*MPW)).  K%32==0.
// ---------------------------------------------------------------------------
template <int MPW>
__global__ __launch_bounds__(256)
void gemm_bf16_wmma(const u16* __restrict__ A, int lda,
                    const u16* __restrict__ Bw, int ldb,
                    float* __restrict__ C, int ldc,
                    const float* __restrict__ addC, int addld, long addoff,
                    const float* __restrict__ bias1,
                    const float* __restrict__ bias2,
                    int K)
{
  constexpr int AROWS = 64 * MPW;          // A rows per block tile
  constexpr int BUFW  = (AROWS + 128) * ROWW;
  constexpr int BUFB  = BUFW * 2;

  __shared__ u16 lds[2][BUFW];

  const int tid  = threadIdx.x;
  const int m0   = blockIdx.y * AROWS;
  const int n0   = blockIdx.x * 128;
  const int w    = tid >> 5;          // wave id (wave32)
  const int lane = tid & 31;
  const int mt   = w & 3;             // base m-tile 0..3 (wave owns mt, mt+4, ..)
  const int nh   = w >> 2;            // n-half 0..1 (4 n-tiles each)
  const int l    = lane & 15;
  const int g    = lane >> 4;

  // loader slots: 4 threads per row, 16B per slot
  const int lr = tid >> 2;            // 0..63
  const int sc = tid & 3;             // col segment 0..3

  const unsigned lbase = (unsigned)(unsigned long long)(&lds[0][0]);

  const u16* aRow[MPW];
  unsigned   aOff[MPW];
#pragma unroll
  for (int i = 0; i < MPW; ++i) {
    aRow[i] = A + (long)(m0 + 64 * i + lr) * lda + sc * 8;
    aOff[i] = (unsigned)((64 * i + lr) * ROWB + sc * 16);
  }
  const u16* bRow0 = Bw + (long)(n0 + lr)      * ldb + sc * 8;
  const u16* bRow1 = Bw + (long)(n0 + 64 + lr) * ldb + sc * 8;
  const unsigned bOff0 = (unsigned)(AROWS * ROWB + lr * ROWB + sc * 16);
  const unsigned bOff1 = (unsigned)(AROWS * ROWB + (64 + lr) * ROWB + sc * 16);

  v8f acc[MPW][4] = {};

  const int nk = K >> 5;

  // prologue: stage chunk 0 into buffer 0
#pragma unroll
  for (int i = 0; i < MPW; ++i) async_b128(lbase + aOff[i], aRow[i]);
  async_b128(lbase + bOff0, bRow0);
  async_b128(lbase + bOff1, bRow1);

  int cur = 0;
  for (int kc = 0; kc < nk; ++kc) {
    wait_async0();                 // own wave's async copies for chunk kc done
    __syncthreads();               // all waves' copies visible

    if (kc + 1 < nk) {             // stage chunk kc+1 into the other buffer
      const int k1 = (kc + 1) << 5;
      const unsigned nb = lbase + (unsigned)((cur ^ 1) * BUFB);
#pragma unroll
      for (int i = 0; i < MPW; ++i) async_b128(nb + aOff[i], aRow[i] + k1);
      async_b128(nb + bOff0, bRow0 + k1);
      async_b128(nb + bOff1, bRow1 + k1);
    }

    // ---- compute from buffer `cur` ----
    // A fragment (16x32 bf16): lanes 0-15 hold K 0..7 / 16..23 of row l,
    // lanes 16-31 hold K 8..15 / 24..31  (ISA 7.12.2 layout)
    const u16* buf = &lds[cur][0];
    FragBF fa[MPW];
#pragma unroll
    for (int i = 0; i < MPW; ++i) {
      const u16* ar = buf + ((mt + 4 * i) * 16 + l) * ROWW;
      fa[i].q[0] = *(const uint4*)(ar + g * 8);
      fa[i].q[1] = *(const uint4*)(ar + 16 + g * 8);
    }

#pragma unroll
    for (int j = 0; j < 4; ++j) {
      // B fragment (32x16 bf16): lane holds one output column; lanes 0-15
      // carry K 0..7 / 16..23, lanes 16-31 carry K 8..15 / 24..31
      FragBF fb;
      const u16* br = buf + (AROWS + nh * 64 + j * 16 + l) * ROWW;
      fb.q[0] = *(const uint4*)(br + g * 8);
      fb.q[1] = *(const uint4*)(br + 16 + g * 8);
#pragma unroll
      for (int i = 0; i < MPW; ++i) {
        acc[i][j] = __builtin_amdgcn_wmma_f32_16x16x32_bf16(
            false, fa[i].v, false, fb.v, (short)0, acc[i][j], false, false);
      }
    }

    __syncthreads();               // buffer `cur` free for reuse at kc+2
    cur ^= 1;
  }

  // Epilogue. C/D layout: VGPR r -> row (g*8 + r), col l.
#pragma unroll
  for (int i = 0; i < MPW; ++i) {
#pragma unroll
    for (int j = 0; j < 4; ++j) {
      FragC cj; cj.v = acc[i][j];
      const int ncol = n0 + nh * 64 + j * 16 + l;
      float badd = 0.0f;
      if (bias1) badd += bias1[ncol];
      if (bias2) badd += bias2[ncol];
#pragma unroll
      for (int r = 0; r < 8; ++r) {
        const int mrow = m0 + (mt + 4 * i) * 16 + g * 8 + r;
        float v = cj.f[r] + badd;
        if (addC) v += addC[(long)mrow * addld + addoff + ncol];
        C[(long)mrow * ldc + ncol] = v;
      }
    }
  }
}

// ---------------------------------------------------------------------------
// LSTM cell pointwise update (fp32 math, bf16 h output)
// ---------------------------------------------------------------------------
__global__ __launch_bounds__(256)
void lstm_cell(const float* __restrict__ gates,   // [B,4H]
               float* __restrict__ c,             // [B,H]
               u16* __restrict__ h_bf,            // [B,H]
               u16* __restrict__ hall,            // [B,T,H]
               int t)
{
  const int idx = blockIdx.x * blockDim.x + threadIdx.x;  // B*H threads
  const int b = idx / HH;
  const int j = idx - b * HH;
  const float* gr = gates + (long)b * G4H;
  const float ig = sigm(gr[j]);
  const float fg = sigm(gr[HH + j]);
  const float gg = tanhf(gr[2 * HH + j]);
  const float og = sigm(gr[3 * HH + j]);
  const float cn = fg * c[idx] + ig * gg;
  c[idx] = cn;
  const u16 hb = f2bf(og * tanhf(cn));
  h_bf[idx] = hb;
  hall[((long)b * TT + t) * HH + j] = hb;
}

__global__ void cvt_f32_bf16(const float* __restrict__ src,
                             u16* __restrict__ dst, long n)
{
  long i = blockIdx.x * (long)blockDim.x + threadIdx.x;
  const long stride = (long)gridDim.x * blockDim.x;
  for (; i < n; i += stride) dst[i] = f2bf(src[i]);
}

__global__ void zero_state(float* __restrict__ c, u16* __restrict__ h, int n)
{
  const int i = blockIdx.x * blockDim.x + threadIdx.x;
  if (i < n) { c[i] = 0.0f; h[i] = 0; }
}

// ---------------------------------------------------------------------------
extern "C" void kernel_launch(void* const* d_in, const int* in_sizes, int n_in,
                              void* d_out, int out_size, void* d_ws, size_t ws_size,
                              hipStream_t stream)
{
  (void)in_sizes; (void)n_in; (void)out_size; (void)ws_size;

  const float* x      = (const float*)d_in[0];
  const float* Wih[3] = { (const float*)d_in[1], (const float*)d_in[5], (const float*)d_in[9]  };
  const float* Whh[3] = { (const float*)d_in[2], (const float*)d_in[6], (const float*)d_in[10] };
  const float* bih[3] = { (const float*)d_in[3], (const float*)d_in[7], (const float*)d_in[11] };
  const float* bhh[3] = { (const float*)d_in[4], (const float*)d_in[8], (const float*)d_in[12] };
  const float* fcW    = (const float*)d_in[13];
  const float* fcb    = (const float*)d_in[14];
  float* out          = (float*)d_out;

  // ---- workspace carve-up --------------------------------------------------
  char* ws = (char*)d_ws;
  size_t off = 0;
  auto take = [&](size_t bytes) -> char* {
    char* p = ws + off;
    off += (bytes + 255) & ~(size_t)255;
    return p;
  };
  u16*   x_bf   = (u16*)take((size_t)BT * INN * sizeof(u16));
  u16*   hallA  = (u16*)take((size_t)BT * HH  * sizeof(u16));
  u16*   hallB  = (u16*)take((size_t)BT * HH  * sizeof(u16));
  u16*   wih_bf[3];
  wih_bf[0] = (u16*)take((size_t)G4H * INN * sizeof(u16));
  wih_bf[1] = (u16*)take((size_t)G4H * HH  * sizeof(u16));
  wih_bf[2] = (u16*)take((size_t)G4H * HH  * sizeof(u16));
  u16*   whh_bf[3];
  for (int i = 0; i < 3; ++i) whh_bf[i] = (u16*)take((size_t)G4H * HH * sizeof(u16));
  u16*   fcw_bf = (u16*)take((size_t)OUTN * HH * sizeof(u16));
  float* xg     = (float*)take((size_t)BT * G4H * sizeof(float));
  float* gates  = (float*)take((size_t)BB * G4H * sizeof(float));
  u16*   h_bf   = (u16*)take((size_t)BB * HH * sizeof(u16));
  float* cst    = (float*)take((size_t)BB * HH * sizeof(float));

  // ---- fp32 -> bf16 staging ------------------------------------------------
  auto cvt = [&](const float* s, u16* d, long n) {
    int blocks = (int)((n + 255) / 256);
    if (blocks > 2048) blocks = 2048;
    cvt_f32_bf16<<<blocks, 256, 0, stream>>>(s, d, n);
  };
  cvt(x, x_bf, (long)BT * INN);
  cvt(Wih[0], wih_bf[0], (long)G4H * INN);
  cvt(Wih[1], wih_bf[1], (long)G4H * HH);
  cvt(Wih[2], wih_bf[2], (long)G4H * HH);
  for (int i = 0; i < 3; ++i) cvt(Whh[i], whh_bf[i], (long)G4H * HH);
  cvt(fcW, fcw_bf, (long)OUTN * HH);

  // ---- layers --------------------------------------------------------------
  const u16* inA = x_bf;
  int din = INN;
  for (int li = 0; li < 3; ++li) {
    u16* outH = (li & 1) ? hallB : hallA;

    // xg = inA * Wih^T + bih + bhh   ->  [BT, 4H] fp32   (128x128 tiles)
    gemm_bf16_wmma<2><<<dim3(G4H / 128, BT / 128), 256, 0, stream>>>(
        inA, din, wih_bf[li], din, xg, G4H,
        nullptr, 0, 0, bih[li], bhh[li], din);

    zero_state<<<(BB * HH) / 256, 256, 0, stream>>>(cst, h_bf, BB * HH);

    for (int t = 0; t < TT; ++t) {
      // gates = h * Whh^T + xg[:, t, :]   (64x128 tiles, M=64)
      gemm_bf16_wmma<1><<<dim3(G4H / 128, 1), 256, 0, stream>>>(
          h_bf, HH, whh_bf[li], HH, gates, G4H,
          xg, TT * G4H, (long)t * G4H, nullptr, nullptr, HH);
      lstm_cell<<<(BB * HH) / 256, 256, 0, stream>>>(gates, cst, h_bf, outH, t);
    }
    inA = outH;
    din = HH;
  }

  // ---- FC head: logits = h_all * fcW^T + fcb -> d_out [BT, OUT] fp32 -------
  gemm_bf16_wmma<2><<<dim3(OUTN / 128, BT / 128), 256, 0, stream>>>(
      inA, HH, fcw_bf, HH, out, OUTN,
      nullptr, 0, 0, fcb, nullptr, HH);
}
